// CachedVideoAttention_17901423690142
// MI455X (gfx1250) — matrix operand order repaired and checked by reading, not compiled, not run on hardware
//
#include <hip/hip_runtime.h>
#include <cmath>

// ---------------------------------------------------------------------------
// CDNA5 (gfx1250) bf16 WMMA types & fragment helpers
// ---------------------------------------------------------------------------
typedef __attribute__((ext_vector_type(16))) __bf16 v16bf;
typedef __attribute__((ext_vector_type(8)))  float  v8f;

union FragU { v16bf v; uint4 q[2]; };

// A-matrix 16x32 bf16 fragment as loaded here (contiguous k order, same as load_frag_b):
//   lane = 16*g + m ; halves j -> K = 16g + j
__device__ __forceinline__ v16bf load_frag_a(const __bf16* base, int ld,
                                             int r0, int k0, int lane) {
  const int g = lane >> 4;
  const __bf16* p = base + (size_t)(r0 + (lane & 15)) * ld + k0;
  FragU f;
  f.q[0] = *(const uint4*)(p + 16 * g);
  f.q[1] = *(const uint4*)(p + 16 * g + 8);
  return f.v;
}

// B-matrix 32x16 bf16 fragment as loaded here (contiguous k order, same as A): lane = 16*g + n ; halves j -> K = 16g + j
__device__ __forceinline__ v16bf load_frag_b(const __bf16* base, int ld,
                                             int n0, int k0, int lane) {
  const int g = lane >> 4;
  const __bf16* p = base + (size_t)(n0 + (lane & 15)) * ld + k0 + 16 * g;
  FragU f;
  f.q[0] = *(const uint4*)(p);
  f.q[1] = *(const uint4*)(p + 8);
  return f.v;
}

__device__ __forceinline__ v8f wmma_bf(v16bf a, v16bf b, v8f c) {
  return __builtin_amdgcn_wmma_f32_16x16x32_bf16(
      false, a, false, b, (short)0, c, false, false);
}

__device__ __forceinline__ float half_reduce_sum(float v) {
  v += __shfl_xor(v, 1); v += __shfl_xor(v, 2);
  v += __shfl_xor(v, 4); v += __shfl_xor(v, 8);
  return v;
}
__device__ __forceinline__ float half_reduce_max(float v) {
  v = fmaxf(v, __shfl_xor(v, 1)); v = fmaxf(v, __shfl_xor(v, 2));
  v = fmaxf(v, __shfl_xor(v, 4)); v = fmaxf(v, __shfl_xor(v, 8));
  return v;
}

// ---------------------------------------------------------------------------
// CDNA5 async global->LDS copy (ASYNCcnt-tracked), via inline asm
// ---------------------------------------------------------------------------
__device__ __forceinline__ uint32_t lds_off(const void* p) {
  return (uint32_t)(size_t)(__attribute__((address_space(3))) const void*)p;
}
__device__ __forceinline__ void async_b128(uint32_t dsoff, const void* gaddr) {
  asm volatile("global_load_async_to_lds_b128 %0, %1, off"
               :: "v"(dsoff), "v"(gaddr) : "memory");
}
__device__ __forceinline__ void wait_async0() {
  asm volatile("s_wait_asynccnt 0x0" ::: "memory");
}

// ---------------------------------------------------------------------------
// Problem constants (shapes fixed by the reference)
// ---------------------------------------------------------------------------
#define B_    2
#define S_    2048
#define D_    1024
#define H_    16
#define DH_   64
#define SC_   2048
#define ST_   (SC_ + S_)          // 4096 total keys
#define ROWS_ (B_ * S_)           // 4096 token rows
#define KC_   64                  // keys per staged LDS chunk

// ---------------------------------------------------------------------------
// Conversion kernels (fp32 -> bf16 staging)
// ---------------------------------------------------------------------------
__global__ void cvt_f32_bf16_kernel(const float* __restrict__ src,
                                    __bf16* __restrict__ dst, int n) {
  int i = (blockIdx.x * blockDim.x + threadIdx.x) * 4;
  if (i < n) {
    float4 v = *(const float4*)(src + i);
    dst[i + 0] = (__bf16)v.x; dst[i + 1] = (__bf16)v.y;
    dst[i + 2] = (__bf16)v.z; dst[i + 3] = (__bf16)v.w;
  }
}

// k_cache [b,h,SC,dh] -> k_full [b,h,ST,dh] rows [0,SC)
__global__ void cvt_kcache_kernel(const float* __restrict__ src,
                                  __bf16* __restrict__ kf) {
  int i = (blockIdx.x * blockDim.x + threadIdx.x) * 4;
  int bh  = i >> 17;                // / (SC_*DH_)
  int rem = i & (SC_ * DH_ - 1);
  float4 v = *(const float4*)(src + i);
  __bf16* d = kf + (size_t)bh * ST_ * DH_ + rem;
  d[0] = (__bf16)v.x; d[1] = (__bf16)v.y; d[2] = (__bf16)v.z; d[3] = (__bf16)v.w;
}

// v_cache [b,h,SC,dh] -> v_t [b,h,dh,ST] cols [0,SC)  (transposed scatter)
__global__ void cvt_vcache_t_kernel(const float* __restrict__ src,
                                    __bf16* __restrict__ vt) {
  int i = blockIdx.x * blockDim.x + threadIdx.x;
  int bh   = i >> 17;
  int rem  = i & (SC_ * DH_ - 1);
  int sidx = rem >> 6;
  int c    = rem & 63;
  vt[((size_t)bh * DH_ + c) * ST_ + sidx] = (__bf16)src[i];
}

// ---------------------------------------------------------------------------
// Kernel 1: qkv = x @ W_qkv^T with fused per-head RMSNorm.
// Fully-unrolled 2-stage register pipeline (no array-rotation copies).
// grid (ROWS_/128, 48), block 256 (8 waves), wave tile 16x64.
// ---------------------------------------------------------------------------
__global__ __launch_bounds__(256) void qkv_rmsnorm_kernel(
    const __bf16* __restrict__ Xb, const __bf16* __restrict__ Wb,
    const float* __restrict__ scale_q, const float* __restrict__ scale_k,
    __bf16* __restrict__ q_ws, __bf16* __restrict__ kf,
    __bf16* __restrict__ vt) {
  const int lane = threadIdx.x & 31;
  const int wave = threadIdx.x >> 5;
  const int row0 = blockIdx.x * 128 + wave * 16;
  const int chunk = blockIdx.y;                    // 0..47
  const int col0 = chunk * 64;
  const int g = lane >> 4, nl = lane & 15;

  v8f acc[4] = {{}, {}, {}, {}};
  v16bf a0 = load_frag_a(Xb, D_, row0, 0, lane);
  v16bf b0[4];
#pragma unroll
  for (int t = 0; t < 4; ++t) b0[t] = load_frag_b(Wb, D_, col0 + t * 16, 0, lane);

#pragma unroll
  for (int k0 = 0; k0 < D_; k0 += 64) {
    __builtin_prefetch(Xb + (size_t)(row0 + nl) * D_ + k0 + 128, 0, 3);
    v16bf a1 = load_frag_a(Xb, D_, row0, k0 + 32, lane);
    v16bf b1[4];
#pragma unroll
    for (int t = 0; t < 4; ++t)
      b1[t] = load_frag_b(Wb, D_, col0 + t * 16, k0 + 32, lane);
#pragma unroll
    for (int t = 0; t < 4; ++t) acc[t] = wmma_bf(a0, b0[t], acc[t]);
    if (k0 + 64 < D_) {
      a0 = load_frag_a(Xb, D_, row0, k0 + 64, lane);
#pragma unroll
      for (int t = 0; t < 4; ++t)
        b0[t] = load_frag_b(Wb, D_, col0 + t * 16, k0 + 64, lane);
    }
#pragma unroll
    for (int t = 0; t < 4; ++t) acc[t] = wmma_bf(a1, b1[t], acc[t]);
  }

  // Fused RMSNorm over dh=64 for q (chunk<16) and k (16<=chunk<32)
  if (chunk < 32) {
    const float* sc = (chunk < 16) ? scale_q : scale_k;
    float sv[4];
#pragma unroll
    for (int t = 0; t < 4; ++t) sv[t] = sc[t * 16 + nl];
#pragma unroll
    for (int r = 0; r < 8; ++r) {
      float ss = 0.f;
#pragma unroll
      for (int t = 0; t < 4; ++t) { float x = acc[t][r]; ss += x * x; }
      ss = half_reduce_sum(ss);
      float inv = 1.0f / (sqrtf(ss * (1.0f / 64.0f)) + 1e-6f);
#pragma unroll
      for (int t = 0; t < 4; ++t) acc[t][r] = acc[t][r] * inv * sv[t];
    }
  }

  const int bidx = row0 >> 11;
  const int srow = row0 & (S_ - 1);
  if (chunk < 16) {                  // Q -> q_ws [b,h,s,dh]
    __bf16* dst = q_ws + (((size_t)(bidx * H_ + chunk)) * S_ + srow) * DH_;
#pragma unroll
    for (int t = 0; t < 4; ++t)
#pragma unroll
      for (int r = 0; r < 8; ++r)
        dst[(size_t)(r + 8 * g) * DH_ + t * 16 + nl] = (__bf16)acc[t][r];
  } else if (chunk < 32) {           // K -> k_full rows [SC_, ST_)
    __bf16* dst = kf + (((size_t)(bidx * H_ + (chunk - 16))) * ST_ + SC_ + srow) * DH_;
#pragma unroll
    for (int t = 0; t < 4; ++t)
#pragma unroll
      for (int r = 0; r < 8; ++r)
        dst[(size_t)(r + 8 * g) * DH_ + t * 16 + nl] = (__bf16)acc[t][r];
  } else {                           // V -> v_t [b,h,dh,ST] cols [SC_, ST_)
    __bf16* dst = vt + ((size_t)(bidx * H_ + (chunk - 32))) * DH_ * ST_;
#pragma unroll
    for (int t = 0; t < 4; ++t)
#pragma unroll
      for (int r = 0; r < 8; ++r)
        dst[(size_t)(t * 16 + nl) * ST_ + SC_ + srow + r + 8 * g] =
            (__bf16)acc[t][r];
  }
}

// ---------------------------------------------------------------------------
// Kernel 2: flash attention per (b,h). grid (B_*H_, S_/256), block 256.
// 32 queries per wave (two 16-row tiles sharing every K/V fragment).
// K/V tiles double-buffered in LDS via global_load_async_to_lds_b128 with
// compile-time buffer indices (key loop unrolled by 2).
// ---------------------------------------------------------------------------
__global__ __launch_bounds__(256) void flash_attn_kernel(
    const __bf16* __restrict__ q_ws, const __bf16* __restrict__ kf,
    const __bf16* __restrict__ vt, __bf16* __restrict__ attn) {
  __shared__ __bf16 Kbuf[2][KC_ * DH_];     // [key][dh]   8 KB each
  __shared__ __bf16 Vbuf[2][DH_ * KC_];     // [dh][key]   8 KB each
  __shared__ __bf16 Pbuf[8][2 * 16 * 32];   // per-wave P tiles (2 x 16x32)

  const int tid = threadIdx.x;
  const int lane = tid & 31;
  const int wave = tid >> 5;
  const int bh = blockIdx.x;
  const int q0 = blockIdx.y * 256 + wave * 32;
  const int g = lane >> 4, nl = lane & 15;

  const __bf16* qbase = q_ws + (size_t)bh * S_ * DH_;
  const __bf16* kbase = kf + (size_t)bh * ST_ * DH_;
  const __bf16* vbase = vt + (size_t)bh * DH_ * ST_;

  v16bf qa[2][2];
#pragma unroll
  for (int mh = 0; mh < 2; ++mh) {
    qa[mh][0] = load_frag_a(qbase, DH_, q0 + mh * 16, 0, lane);
    qa[mh][1] = load_frag_a(qbase, DH_, q0 + mh * 16, 32, lane);
  }

  float mrow[2][8], lrow[2][8];
  v8f o[2][4] = {{{}, {}, {}, {}}, {{}, {}, {}, {}}};
#pragma unroll
  for (int mh = 0; mh < 2; ++mh)
#pragma unroll
    for (int r = 0; r < 8; ++r) { mrow[mh][r] = -1e30f; lrow[mh][r] = 0.f; }

  __bf16* pst = &Pbuf[wave][0];

  // stage one 64-key chunk: 512 x 16B segments each for K and Vt
  auto stage = [&](int buf, int kt) {
#pragma unroll
    for (int i = 0; i < 2; ++i) {
      int s = tid + i * 256;          // 0..511
      int row = s >> 3;               // 0..63
      int part = (s & 7) * 8;         // element offset, 8 bf16 = 16B
      async_b128(lds_off(&Kbuf[buf][row * DH_ + part]),
                 kbase + (size_t)(kt + row) * DH_ + part);
      async_b128(lds_off(&Vbuf[buf][row * KC_ + part]),
                 vbase + (size_t)row * ST_ + kt + part);
    }
  };

  // process one staged 64-key chunk
  auto compute_chunk = [&](const __bf16* kb, const __bf16* vb) {
#pragma unroll
    for (int sub = 0; sub < 2; ++sub) {           // 2 x 32 keys
      v16bf k00 = load_frag_b(kb, DH_, sub * 32, 0, lane);
      v16bf k01 = load_frag_b(kb, DH_, sub * 32, 32, lane);
      v16bf k10 = load_frag_b(kb, DH_, sub * 32 + 16, 0, lane);
      v16bf k11 = load_frag_b(kb, DH_, sub * 32 + 16, 32, lane);
      v16bf vfr[4];
#pragma unroll
      for (int t = 0; t < 4; ++t)
        vfr[t] = load_frag_b(vb, KC_, t * 16, sub * 32, lane);

#pragma unroll
      for (int mh = 0; mh < 2; ++mh) {            // two 16-query tiles
        v8f s0 = {}, s1 = {};
        s0 = wmma_bf(qa[mh][0], k00, s0);
        s0 = wmma_bf(qa[mh][1], k01, s0);
        s1 = wmma_bf(qa[mh][0], k10, s1);
        s1 = wmma_bf(qa[mh][1], k11, s1);

        __bf16* pw = pst + mh * (16 * 32);
#pragma unroll
        for (int r = 0; r < 8; ++r) {
          float mx = half_reduce_max(fmaxf(s0[r], s1[r]));
          float mnew = fmaxf(mrow[mh][r], mx);
          float alpha = __expf(mrow[mh][r] - mnew);
          float p0 = __expf(s0[r] - mnew);
          float p1 = __expf(s1[r] - mnew);
          lrow[mh][r] = lrow[mh][r] * alpha + half_reduce_sum(p0 + p1);
          mrow[mh][r] = mnew;
#pragma unroll
          for (int t = 0; t < 4; ++t) o[mh][t][r] *= alpha;
          const int m = r + 8 * g;
          pw[m * 32 + nl]      = (__bf16)p0;
          pw[m * 32 + 16 + nl] = (__bf16)p1;
        }
        v16bf pa = load_frag_a(pw, 32, 0, 0, lane);
#pragma unroll
        for (int t = 0; t < 4; ++t) o[mh][t] = wmma_bf(pa, vfr[t], o[mh][t]);
      }
    }
  };

  stage(0, 0);
  wait_async0();
  __syncthreads();

  for (int kt = 0; kt < ST_; kt += 2 * KC_) {
    stage(1, kt + KC_);                      // fill buf1 while computing buf0
    compute_chunk(&Kbuf[0][0], &Vbuf[0][0]);
    wait_async0();
    __syncthreads();
    if (kt + 2 * KC_ < ST_) stage(0, kt + 2 * KC_);
    compute_chunk(&Kbuf[1][0], &Vbuf[1][0]);
    wait_async0();
    __syncthreads();
  }

  // epilogue: divide by l, write bf16 into [b, s, h*64 + c]
  const int bidx = bh >> 4, hh = bh & 15;
  __bf16* dst = attn + ((size_t)(bidx * S_ + q0)) * D_ + hh * DH_;
#pragma unroll
  for (int mh = 0; mh < 2; ++mh)
#pragma unroll
    for (int r = 0; r < 8; ++r) {
      float invl = 1.0f / lrow[mh][r];
#pragma unroll
      for (int t = 0; t < 4; ++t)
        dst[(size_t)(mh * 16 + r + 8 * g) * D_ + t * 16 + nl] =
            (__bf16)(o[mh][t][r] * invl);
    }
}

// ---------------------------------------------------------------------------
// Kernel 3: out = attn @ W_o^T (fp32 output), fully-unrolled 2-stage pipeline.
// ---------------------------------------------------------------------------
__global__ __launch_bounds__(256) void out_proj_kernel(
    const __bf16* __restrict__ Ab, const __bf16* __restrict__ Wob,
    float* __restrict__ Out) {
  const int lane = threadIdx.x & 31;
  const int wave = threadIdx.x >> 5;
  const int row0 = blockIdx.x * 128 + wave * 16;
  const int col0 = blockIdx.y * 64;
  const int g = lane >> 4, nl = lane & 15;

  v8f acc[4] = {{}, {}, {}, {}};
  v16bf a0 = load_frag_a(Ab, D_, row0, 0, lane);
  v16bf b0[4];
#pragma unroll
  for (int t = 0; t < 4; ++t) b0[t] = load_frag_b(Wob, D_, col0 + t * 16, 0, lane);

#pragma unroll
  for (int k0 = 0; k0 < D_; k0 += 64) {
    __builtin_prefetch(Ab + (size_t)(row0 + nl) * D_ + k0 + 128, 0, 3);
    v16bf a1 = load_frag_a(Ab, D_, row0, k0 + 32, lane);
    v16bf b1[4];
#pragma unroll
    for (int t = 0; t < 4; ++t)
      b1[t] = load_frag_b(Wob, D_, col0 + t * 16, k0 + 32, lane);
#pragma unroll
    for (int t = 0; t < 4; ++t) acc[t] = wmma_bf(a0, b0[t], acc[t]);
    if (k0 + 64 < D_) {
      a0 = load_frag_a(Ab, D_, row0, k0 + 64, lane);
#pragma unroll
      for (int t = 0; t < 4; ++t)
        b0[t] = load_frag_b(Wob, D_, col0 + t * 16, k0 + 64, lane);
    }
#pragma unroll
    for (int t = 0; t < 4; ++t) acc[t] = wmma_bf(a1, b1[t], acc[t]);
  }

#pragma unroll
  for (int t = 0; t < 4; ++t)
#pragma unroll
    for (int r = 0; r < 8; ++r)
      Out[(size_t)(row0 + r + 8 * g) * D_ + col0 + t * 16 + nl] = acc[t][r];
}

// ---------------------------------------------------------------------------
// Host launcher
// ---------------------------------------------------------------------------
extern "C" void kernel_launch(void* const* d_in, const int* in_sizes, int n_in,
                              void* d_out, int out_size, void* d_ws,
                              size_t ws_size, hipStream_t stream) {
  (void)in_sizes; (void)n_in; (void)out_size; (void)ws_size;
  const float* x    = (const float*)d_in[0];
  const float* kc   = (const float*)d_in[1];
  const float* vc   = (const float*)d_in[2];
  const float* wqkv = (const float*)d_in[3];
  const float* wo   = (const float*)d_in[4];
  const float* sq   = (const float*)d_in[5];
  const float* sk   = (const float*)d_in[6];
  float* out = (float*)d_out;

  size_t off = 0;
  auto carve = [&](size_t bytes) -> char* {
    char* p = (char*)d_ws + off;
    off += (bytes + 255) & ~(size_t)255;
    return p;
  };
  __bf16* xb    = (__bf16*)carve((size_t)ROWS_ * D_ * 2);
  __bf16* wqkvb = (__bf16*)carve((size_t)3 * D_ * D_ * 2);
  __bf16* wob   = (__bf16*)carve((size_t)D_ * D_ * 2);
  __bf16* qws   = (__bf16*)carve((size_t)B_ * H_ * S_ * DH_ * 2);
  __bf16* kfull = (__bf16*)carve((size_t)B_ * H_ * ST_ * DH_ * 2);
  __bf16* vtw   = (__bf16*)carve((size_t)B_ * H_ * DH_ * ST_ * 2);
  __bf16* attnw = (__bf16*)carve((size_t)ROWS_ * D_ * 2);

  {
    int n = ROWS_ * D_;
    cvt_f32_bf16_kernel<<<n / 1024, 256, 0, stream>>>(x, xb, n);
    n = 3 * D_ * D_;
    cvt_f32_bf16_kernel<<<n / 1024, 256, 0, stream>>>(wqkv, wqkvb, n);
    n = D_ * D_;
    cvt_f32_bf16_kernel<<<n / 1024, 256, 0, stream>>>(wo, wob, n);
    n = B_ * H_ * SC_ * DH_;
    cvt_kcache_kernel<<<n / 1024, 256, 0, stream>>>(kc, kfull);
    cvt_vcache_t_kernel<<<n / 256, 256, 0, stream>>>(vc, vtw);
  }

  qkv_rmsnorm_kernel<<<dim3(ROWS_ / 128, 48), 256, 0, stream>>>(
      xb, wqkvb, sq, sk, qws, kfull, vtw);

  flash_attn_kernel<<<dim3(B_ * H_, S_ / 256), 256, 0, stream>>>(
      qws, kfull, vtw, attnw);

  out_proj_kernel<<<dim3(ROWS_ / 128, D_ / 64), 256, 0, stream>>>(
      attnw, wob, out);
}
